// WanSelfAttention_55714315763718
// MI455X (gfx1250) — matrix operand order repaired and checked by reading, not compiled, not run on hardware
//
#include <hip/hip_runtime.h>

// ---------------- problem constants ----------------
#define B_    2
#define S_    3072
#define DIM_  1536
#define NH_   12
#define D_    128       // head dim
#define KDIM_ 1536      // GEMM K
#define MROWS_ (B_*S_)  // 6144

typedef _Float16 v8h  __attribute__((ext_vector_type(8)));
typedef _Float16 v16h __attribute__((ext_vector_type(16)));
typedef float    v8f  __attribute__((ext_vector_type(8)));

// ---------------- WMMA helpers (gfx1250, wave32) ----------------
// A-matrix 16x32 f16 (ISA 7.12.2): lane L -> row M=L%16, h=L/16;
// element j -> K = (j&7) + 16*(j>>3) + 8*h   (two 16-byte chunks)
__device__ __forceinline__ v16h frag_a(const _Float16* base, int h) {
  v8h lo = *reinterpret_cast<const v8h*>(base + 8 * h);       // K = 8h .. 8h+7
  v8h hi = *reinterpret_cast<const v8h*>(base + 16 + 8 * h);  // K = 16+8h .. 23+8h
  return __builtin_shufflevector(lo, hi, 0,1,2,3,4,5,6,7,8,9,10,11,12,13,14,15);
}
// B-matrix 32x16 f16 (per SWMMAC B pattern): lane L -> col N=L%16, h=L/16;
// element j -> K = (j&7) + 16*(j>>3) + 8*h   (two 16-byte chunks)
__device__ __forceinline__ v16h frag_b(const _Float16* base, int h) {
  v8h lo = *reinterpret_cast<const v8h*>(base + 8 * h);
  v8h hi = *reinterpret_cast<const v8h*>(base + 16 + 8 * h);
  return __builtin_shufflevector(lo, hi, 0,1,2,3,4,5,6,7,8,9,10,11,12,13,14,15);
}
__device__ __forceinline__ v8f wmma16(v16h a, v16h b, v8f c) {
  return __builtin_amdgcn_wmma_f32_16x16x32_f16(false, a, false, b, (short)0, c, false, false);
}
// C/D layout: VGPR r, lane L -> row M = r + 8*(L>>4), col N = L%16

// ---------------- CDNA5 async global->LDS (ASYNCcnt) ----------------
// GV mode: 64-bit global address in VGPR pair, LDS byte address in VGPR.
__device__ __forceinline__ void async_ld_b128(const _Float16* ldsDst, const _Float16* gsrc) {
  unsigned lds_off = (unsigned)(unsigned long long)ldsDst;   // LDS aperture: low 32 bits
  unsigned long long gaddr = (unsigned long long)gsrc;
  asm volatile("global_load_async_to_lds_b128 %0, %1, off"
               :: "v"(lds_off), "v"(gaddr) : "memory");
}
__device__ __forceinline__ void wait_async0() {
  asm volatile("s_wait_asynccnt 0x0" ::: "memory");
}

// ---------------- GEMM: Y[M,N] = X[M,K] * W[N,K]^T + bias ----------------
// M=6144, N=1536, K=1536 fixed. 128x128 tile/WG, 8 waves -> 64x32 per wave.
#define TK_  64
#define LDT_ 72   // padded LDS row stride in halves (144 B, multiple of 16 B)

__global__ __launch_bounds__(256) void gemm_kernel(
    float* __restrict__ Y, const float* __restrict__ X,
    const float* __restrict__ W, const float* __restrict__ bias) {
  __shared__ _Float16 ldsA[128 * LDT_];
  __shared__ _Float16 ldsB[128 * LDT_];
  const int tid  = threadIdx.x;
  const int lane = tid & 31;
  const int wave = tid >> 5;
  const int h    = lane >> 4;
  const int lm   = lane & 15;
  const int n0   = blockIdx.x * 128;
  const int m0   = blockIdx.y * 128;
  const int wm   = (wave & 1) * 64;   // wave row base inside tile
  const int wn   = (wave >> 1) * 32;  // wave col base inside tile

  v8f acc[4][2];
#pragma unroll
  for (int i = 0; i < 4; ++i)
#pragma unroll
    for (int j = 0; j < 2; ++j) { v8f z = {}; acc[i][j] = z; }

  for (int k0 = 0; k0 < KDIM_; k0 += TK_) {
    __syncthreads();
    // stage A tile 128x64 (f32 -> f16), 2048 float4 chunks
#pragma unroll
    for (int it = 0; it < 8; ++it) {
      int c  = tid + it * 256;
      int r  = c >> 4;          // 16 float4 chunks per row
      int kv = c & 15;
      float4 v = *reinterpret_cast<const float4*>(&X[(size_t)(m0 + r) * KDIM_ + k0 + kv * 4]);
      _Float16* d = &ldsA[r * LDT_ + kv * 4];
      d[0] = (_Float16)v.x; d[1] = (_Float16)v.y; d[2] = (_Float16)v.z; d[3] = (_Float16)v.w;
    }
    // stage B tile (rows = output cols n, contiguous K): ldsB[n][k]
#pragma unroll
    for (int it = 0; it < 8; ++it) {
      int c  = tid + it * 256;
      int r  = c >> 4;
      int kv = c & 15;
      float4 v = *reinterpret_cast<const float4*>(&W[(size_t)(n0 + r) * KDIM_ + k0 + kv * 4]);
      _Float16* d = &ldsB[r * LDT_ + kv * 4];
      d[0] = (_Float16)v.x; d[1] = (_Float16)v.y; d[2] = (_Float16)v.z; d[3] = (_Float16)v.w;
    }
    __syncthreads();
#pragma unroll
    for (int kc = 0; kc < 2; ++kc) {
      v16h bf[2];
#pragma unroll
      for (int j = 0; j < 2; ++j)
        bf[j] = frag_b(&ldsB[(wn + j * 16 + lm) * LDT_ + kc * 32], h);
#pragma unroll
      for (int i = 0; i < 4; ++i) {
        v16h af = frag_a(&ldsA[(wm + i * 16 + lm) * LDT_ + kc * 32], h);
#pragma unroll
        for (int j = 0; j < 2; ++j)
          acc[i][j] = wmma16(af, bf[j], acc[i][j]);
      }
    }
  }
#pragma unroll
  for (int i = 0; i < 4; ++i)
#pragma unroll
    for (int j = 0; j < 2; ++j) {
      int col  = n0 + wn + j * 16 + lm;
      float bv = bias[col];
      int rowb = m0 + wm + i * 16 + h * 8;
#pragma unroll
      for (int r = 0; r < 8; ++r)
        Y[(size_t)(rowb + r) * DIM_ + col] = acc[i][j][r] + bv;
    }
}

// ---------------- bias already in Y; RMSNorm over 1536 + gain + RoPE ----------------
// out f16 layout: [b][h][s][d]
__global__ __launch_bounds__(256) void rmsrope_kernel(
    _Float16* __restrict__ out, const float* __restrict__ Y,
    const float* __restrict__ g, const float* __restrict__ freqs,
    const int* __restrict__ grid_sizes) {
  const int s = blockIdx.x, b = blockIdx.y, tid = threadIdx.x;
  const float* y = Y + (size_t)(b * S_ + s) * DIM_;
  __shared__ float red[256];
  float acc = 0.f;
#pragma unroll
  for (int i = 0; i < DIM_ / 256; ++i) {
    float v = y[tid + i * 256];
    acc += v * v;
  }
  red[tid] = acc;
  __syncthreads();
  for (int off = 128; off > 0; off >>= 1) {
    if (tid < off) red[tid] += red[tid + off];
    __syncthreads();
  }
  const float rinv = rsqrtf(red[0] * (1.0f / (float)DIM_) + 1e-6f);
  const int gh = grid_sizes[b * 3 + 1], gw = grid_sizes[b * 3 + 2];
  const int iF = s / (gh * gw);
  int rem = s - iF * gh * gw;
  const int iH = rem / gw, iW = rem - (rem / gw) * gw;
  // C=64 rope pairs per head; s1=21, s0=22 -> c<22: f, c<43: h, else w
#pragma unroll
  for (int it = 0; it < 3; ++it) {           // 12*64 = 768 pairs / 256 threads
    int p  = tid + it * 256;
    int hh = p >> 6;
    int c  = p & 63;
    int row = (c < 22) ? iF : ((c < 43) ? iH : iW);
    float ang = freqs[row * 64 + c];
    float sn, cs;
    __sincosf(ang, &sn, &cs);
    int d0  = hh * 128 + 2 * c;
    float x0 = y[d0] * rinv * g[d0];
    float x1 = y[d0 + 1] * rinv * g[d0 + 1];
    _Float16* o = out + (((size_t)(b * NH_ + hh) * S_ + s) * D_ + 2 * c);
    o[0] = (_Float16)(x0 * cs - x1 * sn);
    o[1] = (_Float16)(x0 * sn + x1 * cs);
  }
}

// ---------------- V: f32 [b,s,h*128+d] -> f16 [b,h,d,s] ----------------
__global__ __launch_bounds__(256) void vtrans_kernel(
    _Float16* __restrict__ Vt, const float* __restrict__ Y) {
  size_t i = (size_t)blockIdx.x * 256 + threadIdx.x;
  if (i >= (size_t)MROWS_ * DIM_) return;
  int c  = (int)(i % DIM_);
  size_t bs = i / DIM_;
  int s = (int)(bs % S_), b = (int)(bs / S_);
  int hh = c >> 7, d = c & 127;
  Vt[(((size_t)(b * NH_ + hh) * D_ + d) * S_ + s)] = (_Float16)Y[i];
}

// ---------------- flash attention ----------------
// grid: (S/128, B*NH); 8 waves x 16 query rows; 64-key tiles; online softmax.
// K/V^T tiles double-buffered in LDS, filled with global_load_async_to_lds_b128
// (ASYNCcnt) so DMA of tile t+1 overlaps WMMA+softmax of tile t.
#define LKK_ 136   // K tile row stride (halves), 272 B
#define LVV_ 72    // V^T / P row stride (halves), 144 B

__device__ __forceinline__ void stage_tiles_async(
    _Float16* ldsKbuf, _Float16* ldsVbuf,
    const _Float16* Kg,  // &Kh[bh][kt][0]   rows: 64 x 128 halves
    const _Float16* Vg,  // &Vt[bh][0][kt]   rows: 128 x 64 halves (row stride S_)
    int tid) {
#pragma unroll
  for (int it = 0; it < 4; ++it) {
    int c = tid + it * 256;
    int r = c >> 4, c8 = c & 15;
    async_ld_b128(&ldsKbuf[r * LKK_ + c8 * 8], &Kg[(size_t)r * D_ + c8 * 8]);
  }
#pragma unroll
  for (int it = 0; it < 4; ++it) {
    int c = tid + it * 256;
    int r = c >> 3, c8 = c & 7;
    async_ld_b128(&ldsVbuf[r * LVV_ + c8 * 8], &Vg[(size_t)r * S_ + c8 * 8]);
  }
}

__global__ __launch_bounds__(256) void flash_kernel(
    float* __restrict__ Oa,                // f32 [b,s, h*128+d]
    const _Float16* __restrict__ Qh,       // [bh][s][d]
    const _Float16* __restrict__ Kh,       // [bh][s][d]
    const _Float16* __restrict__ Vt,       // [bh][d][s]
    const int* __restrict__ seq_lens) {
  __shared__ _Float16 ldsK[2][64 * LKK_];
  __shared__ _Float16 ldsV[2][D_ * LVV_];
  __shared__ _Float16 ldsP[8 * 16 * LVV_];
  const int tid  = threadIdx.x;
  const int lane = tid & 31;
  const int wave = tid >> 5;
  const int h    = lane >> 4;
  const int lm   = lane & 15;
  const int q0   = blockIdx.x * 128;
  const int bh   = blockIdx.y;
  const int b    = bh / NH_, head = bh % NH_;
  const int slen = seq_lens[b];
  const float scale = 0.08838834764831845f;   // 1/sqrt(128)

  const _Float16* Kbase = Kh + (size_t)bh * S_ * D_;
  const _Float16* Vbase = Vt + (size_t)bh * D_ * S_;

  // resident Q A-fragments (16 rows x 128 d = 4 K-chunks)
  v16h aq[4];
  {
    const _Float16* qb = Qh + ((size_t)bh * S_ + (q0 + wave * 16 + lm)) * D_;
#pragma unroll
    for (int kc = 0; kc < 4; ++kc) aq[kc] = frag_a(qb + kc * 32, h);
  }
  v8f o[8];
#pragma unroll
  for (int i = 0; i < 8; ++i) { v8f z = {}; o[i] = z; }
  float m[8], l[8];
#pragma unroll
  for (int r = 0; r < 8; ++r) { m[r] = -1e30f; l[r] = 0.f; }

  // prologue: DMA tile 0 into buffer 0
  stage_tiles_async(ldsK[0], ldsV[0], Kbase, Vbase, tid);
  wait_async0();
  __syncthreads();

  const int NT = S_ / 64;
  for (int t = 0; t < NT; ++t) {
    const int kt  = t * 64;
    const int cur = t & 1;
    // prefetch tile t+1 into the other buffer (overlaps with compute below)
    if (t + 1 < NT)
      stage_tiles_async(ldsK[cur ^ 1], ldsV[cur ^ 1],
                        Kbase + (size_t)(kt + 64) * D_, Vbase + (kt + 64), tid);

    const _Float16* lk = ldsK[cur];
    const _Float16* lv = ldsV[cur];

    // scores S[16x64] = Q[16x128] * K^T
    v8f sc[4];
#pragma unroll
    for (int nt = 0; nt < 4; ++nt) {
      v8f a = {};
#pragma unroll
      for (int kc = 0; kc < 4; ++kc) {
        v16h bf = frag_b(&lk[(nt * 16 + lm) * LKK_ + kc * 32], h);
        a = wmma16(aq[kc], bf, a);
      }
      sc[nt] = a;
    }
    // scale + key mask
#pragma unroll
    for (int nt = 0; nt < 4; ++nt) {
      bool valid = (kt + nt * 16 + lm) < slen;
#pragma unroll
      for (int r = 0; r < 8; ++r)
        sc[nt][r] = valid ? sc[nt][r] * scale : -1e30f;
    }
    // online softmax (row = r + 8h, cols across the 16 lanes of this half)
#pragma unroll
    for (int r = 0; r < 8; ++r) {
      float v = fmaxf(fmaxf(sc[0][r], sc[1][r]), fmaxf(sc[2][r], sc[3][r]));
      v = fmaxf(v, __shfl_xor(v, 1));
      v = fmaxf(v, __shfl_xor(v, 2));
      v = fmaxf(v, __shfl_xor(v, 4));
      v = fmaxf(v, __shfl_xor(v, 8));
      float mn = fmaxf(m[r], v);
      float s0 = 0.f;
#pragma unroll
      for (int nt = 0; nt < 4; ++nt) {
        float p = __expf(sc[nt][r] - mn);
        sc[nt][r] = p;
        s0 += p;
      }
      s0 += __shfl_xor(s0, 1);
      s0 += __shfl_xor(s0, 2);
      s0 += __shfl_xor(s0, 4);
      s0 += __shfl_xor(s0, 8);
      float corr = __expf(m[r] - mn);
      l[r] = l[r] * corr + s0;
      m[r] = mn;
#pragma unroll
      for (int nd = 0; nd < 8; ++nd) o[nd][r] *= corr;
    }
    // reshape P (D-layout) -> A-layout via per-wave LDS scratch
    _Float16* pw = &ldsP[wave * 16 * LVV_];
#pragma unroll
    for (int nt = 0; nt < 4; ++nt)
#pragma unroll
      for (int r = 0; r < 8; ++r)
        pw[(r + 8 * h) * LVV_ + nt * 16 + lm] = (_Float16)sc[nt][r];
    v16h pf[2];
#pragma unroll
    for (int kc = 0; kc < 2; ++kc) pf[kc] = frag_a(&pw[lm * LVV_ + kc * 32], h);
    // O[16x128] += P[16x64] * V[64x128]
#pragma unroll
    for (int nd = 0; nd < 8; ++nd)
#pragma unroll
      for (int kc = 0; kc < 2; ++kc) {
        v16h bf = frag_b(&lv[(nd * 16 + lm) * LVV_ + kc * 32], h);
        o[nd] = wmma16(pf[kc], bf, o[nd]);
      }

    // tile t+1 DMA must be complete and all waves done with buffers
    wait_async0();
    __syncthreads();
  }
  // epilogue: normalize and write f32 [b,s,h*128+d]
#pragma unroll
  for (int r = 0; r < 8; ++r) {
    float inv = 1.f / l[r];
    int row = q0 + wave * 16 + r + 8 * h;
#pragma unroll
    for (int nd = 0; nd < 8; ++nd) {
      int col = head * D_ + nd * 16 + lm;
      Oa[((size_t)b * S_ + row) * DIM_ + col] = o[nd][r] * inv;
    }
  }
}

// ---------------- host launch ----------------
extern "C" void kernel_launch(void* const* d_in, const int* in_sizes, int n_in,
                              void* d_out, int out_size, void* d_ws, size_t ws_size,
                              hipStream_t stream) {
  (void)in_sizes; (void)n_in; (void)out_size; (void)ws_size;
  const float* x        = (const float*)d_in[0];
  const int*  seq_lens  = (const int*)d_in[1];
  const int*  grid_sz   = (const int*)d_in[2];
  const float* freqs    = (const float*)d_in[3];
  const float* Wq = (const float*)d_in[4];  const float* bq = (const float*)d_in[5];
  const float* Wk = (const float*)d_in[6];  const float* bk = (const float*)d_in[7];
  const float* Wv = (const float*)d_in[8];  const float* bv = (const float*)d_in[9];
  const float* Wo = (const float*)d_in[10]; const float* bo = (const float*)d_in[11];
  const float* gq = (const float*)d_in[12]; const float* gk = (const float*)d_in[13];
  float* out = (float*)d_out;

  char* ws = (char*)d_ws;
  float*     Yw = (float*)ws;                           // 6144*1536 f32  (37,748,736 B)
  _Float16*  Qh = (_Float16*)(ws + 37748736ull);        // 18,874,368 B
  _Float16*  Kh = (_Float16*)(ws + 56623104ull);        // 18,874,368 B
  _Float16*  Vt = (_Float16*)(ws + 75497472ull);        // 18,874,368 B
  float*     Oa = (float*)(ws + 94371840ull);           // 37,748,736 B  (total ~132 MB)

  dim3 gG(DIM_ / 128, MROWS_ / 128);                    // (12, 48)
  dim3 gR(S_, B_);
  dim3 gF(S_ / 128, B_ * NH_);                          // (24, 24)

  gemm_kernel<<<gG, 256, 0, stream>>>(Yw, x, Wq, bq);
  rmsrope_kernel<<<gR, 256, 0, stream>>>(Qh, Yw, gq, freqs, grid_sz);
  gemm_kernel<<<gG, 256, 0, stream>>>(Yw, x, Wk, bk);
  rmsrope_kernel<<<gR, 256, 0, stream>>>(Kh, Yw, gk, freqs, grid_sz);
  gemm_kernel<<<gG, 256, 0, stream>>>(Yw, x, Wv, bv);
  vtrans_kernel<<<(MROWS_ * DIM_) / 256, 256, 0, stream>>>(Vt, Yw);
  flash_kernel<<<gF, 256, 0, stream>>>(Oa, Qh, Kh, Vt, seq_lens);
  gemm_kernel<<<gG, 256, 0, stream>>>(out, Oa, Wo, bo);
}
